// BlockSparseModel_67216238182776
// MI455X (gfx1250) — matrix operand order repaired and checked by reading, not compiled, not run on hardware
//
#include <hip/hip_runtime.h>

// ---------------------------------------------------------------------------
// BlockSparse MLP for MI455X (gfx1250), wave32, WMMA bf16 path + async LDS.
//   h1 = relu(x  @ (W1*mask1).T + b1)   [8192,4096]
//   h2 = relu(h1 @ (W2*mask2).T + b2)   [8192,4096]
//   y  =      h2 @ Wo.T        + bo     [8192,1024]
// Block tile M=256 x N=32, K-step 32 == one 32x32 mask tile (uniform skip).
// 8 waves/WG, each wave owns 32x32 -> 4 x v_wmma_f32_16x16x32_bf16 per step.
// A tile staged via GLOBAL_LOAD_ASYNC_TO_LDS_B128 (ASYNCcnt), W tile staged
// synchronously with f32->bf16 convert. next-W prefetch via global_prefetch.
// ---------------------------------------------------------------------------

typedef unsigned short u16;
typedef unsigned int   u32;
typedef __attribute__((ext_vector_type(16))) __bf16 v16bf;
typedef __attribute__((ext_vector_type(8)))  float  v8f;

__device__ __forceinline__ u16 cvt_bf16(float f) {
  u32 u = __float_as_uint(f);
  u32 r = u + 0x7FFFu + ((u >> 16) & 1u);   // round-to-nearest-even
  return (u16)(r >> 16);
}

// -------------------------- f32 -> bf16 convert ----------------------------
__global__ void __launch_bounds__(256)
cvt_f32_bf16_kernel(const float* __restrict__ src, u16* __restrict__ dst, int n) {
  int i = (blockIdx.x * 256 + threadIdx.x) * 4;
  if (i >= n) return;
  float4 v = *(const float4*)(src + i);
  u32 p0 = (u32)cvt_bf16(v.x) | ((u32)cvt_bf16(v.y) << 16);
  u32 p1 = (u32)cvt_bf16(v.z) | ((u32)cvt_bf16(v.w) << 16);
  uint2 pk; pk.x = p0; pk.y = p1;
  *(uint2*)(dst + i) = pk;
}

// 64-byte global->LDS async copy (GVS mode: SGPR base + VGPR byte offset).
// INST_OFFSET is added to BOTH the LDS and memory addresses (ISA 08 §4.4).
__device__ __forceinline__ void async_copy_64B(u32 lds_byte, u32 goff_byte,
                                               const u16* gbase) {
  asm volatile(
      "global_load_async_to_lds_b128 %0, %1, %2 offset:0\n\t"
      "global_load_async_to_lds_b128 %0, %1, %2 offset:16\n\t"
      "global_load_async_to_lds_b128 %0, %1, %2 offset:32\n\t"
      "global_load_async_to_lds_b128 %0, %1, %2 offset:48"
      :
      : "v"(lds_byte), "v"(goff_byte), "s"(gbase)
      : "memory");
}

__device__ __forceinline__ void wait_asynccnt0() {
  asm volatile("s_wait_asynccnt 0x0" ::: "memory");
}

// ------------------------------ NT GEMM ------------------------------------
// out[M,N] = act( A[M,K](bf16) @ W[N,K](f32, cvt->bf16 in LDS).T + bias )
template <bool USE_MASK, bool RELU, bool OUT_BF16>
__global__ void __launch_bounds__(256, 2)
gemm_nt_kernel(const u16* __restrict__ A, const float* __restrict__ W,
               const float* __restrict__ bias,
               const unsigned char* __restrict__ mask,
               void* __restrict__ outp, int M, int N, int K) {
  __shared__ alignas(16) u16 As[256][40];   // 80B row stride (bank spread)
  __shared__ alignas(16) u16 Ws[32][40];

  const int tid  = threadIdx.x;
  const int m0   = blockIdx.y * 256;
  const int n0   = blockIdx.x * 32;
  const int KT   = K >> 5;

  const int lane = tid & 31;
  const int wave = tid >> 5;
  const int lh   = lane & 15;
  const int hi   = lane >> 4;            // 0: lanes 0-15, 1: lanes 16-31
  const int mw   = wave * 32;            // wave row offset in tile

  // staging assignments
  // A: 256 rows x 32 bf16 (64B/row) -> one async row per thread
  const u32 lds_a   = (u32)(uintptr_t)&As[tid][0];
  const size_t arow = (size_t)(m0 + tid) * K;
  // W: 32 rows x 32 f32 -> 4 f32 per thread, convert to bf16
  const int wrow = tid >> 3, wseg = (tid & 7) * 4;

  v8f acc00 = {}, acc01 = {}, acc10 = {}, acc11 = {};

  for (int k0 = 0; k0 < K; k0 += 32) {
    if (USE_MASK) {
      // one mask bit per (n-tile, k-tile): uniform across the workgroup
      if (!mask[(size_t)blockIdx.x * KT + (k0 >> 5)]) continue;
    }
    // ---- stage A tile: async global->LDS, 64B per thread ----
    async_copy_64B(lds_a, (u32)((arow + k0) * 2), A);
    // ---- stage W tile: f32 -> bf16 ----
    {
      const float* wsrc = W + (size_t)(n0 + wrow) * K + k0 + wseg;
      float4 wv = *(const float4*)wsrc;
      uint2 pk;
      pk.x = (u32)cvt_bf16(wv.x) | ((u32)cvt_bf16(wv.y) << 16);
      pk.y = (u32)cvt_bf16(wv.z) | ((u32)cvt_bf16(wv.w) << 16);
      *(uint2*)&Ws[wrow][wseg] = pk;
      if (k0 + 32 < K) __builtin_prefetch(wsrc + 32, 0, 3); // global_prefetch
    }
    wait_asynccnt0();          // our async writes landed in LDS
    __syncthreads();

    // ---- fragments per ISA 7.12.2 (wave32) ----
    // A 16x32 bf16: lane<16 holds K0-7,K16-23 of row M=lh; lane>=16 K8-15,K24-31
    v16bf a0, a1, b0, b1;
    {
      uint4* p = (uint4*)&a0;
      p[0] = *(const uint4*)&As[mw + lh][hi * 8];
      p[1] = *(const uint4*)&As[mw + lh][16 + hi * 8];
      p = (uint4*)&a1;
      p[0] = *(const uint4*)&As[mw + 16 + lh][hi * 8];
      p[1] = *(const uint4*)&As[mw + 16 + lh][16 + hi * 8];
      // B 32x16 bf16: lane<16 holds K0-7,K16-23 of col N=lh (= row of W); lane>=16 K8-15,K24-31
      p = (uint4*)&b0;
      p[0] = *(const uint4*)&Ws[lh][hi * 8];
      p[1] = *(const uint4*)&Ws[lh][16 + hi * 8];
      p = (uint4*)&b1;
      p[0] = *(const uint4*)&Ws[16 + lh][hi * 8];
      p[1] = *(const uint4*)&Ws[16 + lh][16 + hi * 8];
    }

    acc00 = __builtin_amdgcn_wmma_f32_16x16x32_bf16(
        false, a0, false, b0, (short)0, acc00, false, false);
    acc01 = __builtin_amdgcn_wmma_f32_16x16x32_bf16(
        false, a0, false, b1, (short)0, acc01, false, false);
    acc10 = __builtin_amdgcn_wmma_f32_16x16x32_bf16(
        false, a1, false, b0, (short)0, acc10, false, false);
    acc11 = __builtin_amdgcn_wmma_f32_16x16x32_bf16(
        false, a1, false, b1, (short)0, acc11, false, false);

    __syncthreads();
  }

  // ---- epilogue: C/D layout: VGPR r -> row base+hi*8+r, col = lh ----
  const int c0 = n0 + lh;
  const int c1 = n0 + 16 + lh;
  const float bv0 = bias[c0];
  const float bv1 = bias[c1];
  const int rm = m0 + mw + hi * 8;
#pragma unroll
  for (int r = 0; r < 8; ++r) {
    float v00 = acc00[r] + bv0, v01 = acc01[r] + bv1;
    float v10 = acc10[r] + bv0, v11 = acc11[r] + bv1;
    if (RELU) {
      v00 = fmaxf(v00, 0.f); v01 = fmaxf(v01, 0.f);
      v10 = fmaxf(v10, 0.f); v11 = fmaxf(v11, 0.f);
    }
    const size_t row0 = (size_t)(rm + r) * N;
    const size_t row1 = (size_t)(rm + 16 + r) * N;
    if (OUT_BF16) {
      u16* out = (u16*)outp;
      out[row0 + c0] = cvt_bf16(v00);
      out[row0 + c1] = cvt_bf16(v01);
      out[row1 + c0] = cvt_bf16(v10);
      out[row1 + c1] = cvt_bf16(v11);
    } else {
      float* out = (float*)outp;
      out[row0 + c0] = v00;
      out[row0 + c1] = v01;
      out[row1 + c0] = v10;
      out[row1 + c1] = v11;
    }
  }
}

// ------------------------------- launcher ----------------------------------
extern "C" void kernel_launch(void* const* d_in, const int* in_sizes, int n_in,
                              void* d_out, int out_size, void* d_ws, size_t ws_size,
                              hipStream_t stream) {
  (void)in_sizes; (void)n_in; (void)out_size; (void)ws_size;
  const int B = 8192, DI = 2048, DH = 4096, DO = 1024;

  const float* x  = (const float*)d_in[0];
  const float* W1 = (const float*)d_in[1];
  const float* b1 = (const float*)d_in[2];
  const float* W2 = (const float*)d_in[3];
  const float* b2 = (const float*)d_in[4];
  const float* Wo = (const float*)d_in[5];
  const float* bo = (const float*)d_in[6];
  const unsigned char* mask1 = (const unsigned char*)d_in[7]; // [DH/32, DI/32] bool
  const unsigned char* mask2 = (const unsigned char*)d_in[8]; // [DH/32, DH/32] bool

  // workspace layout (bf16): x_bf 32MB | h1 64MB | h2 64MB  (160MB total)
  u16* xb = (u16*)d_ws;
  u16* h1 = xb + (size_t)B * DI;
  u16* h2 = h1 + (size_t)B * DH;

  // 1) convert x to bf16
  {
    int n = B * DI;
    cvt_f32_bf16_kernel<<<n / 1024, 256, 0, stream>>>(x, xb, n);
  }
  // 2) layer 1: block-sparse + relu -> bf16
  gemm_nt_kernel<true, true, true>
      <<<dim3(DH / 32, B / 256), 256, 0, stream>>>(xb, W1, b1, mask1, h1, B, DH, DI);
  // 3) layer 2: block-sparse + relu -> bf16
  gemm_nt_kernel<true, true, true>
      <<<dim3(DH / 32, B / 256), 256, 0, stream>>>(h1, W2, b2, mask2, h2, B, DH, DH);
  // 4) output head: dense, f32 out
  gemm_nt_kernel<false, false, false>
      <<<dim3(DO / 32, B / 256), 256, 0, stream>>>(h2, Wo, bo, nullptr, d_out, B, DO, DH);
}
